// SparseModel_21303037788645
// MI455X (gfx1250) — matrix operand order repaired and checked by reading, not compiled, not run on hardware
//
#include <hip/hip_runtime.h>

// ---- CDNA5 WMMA types (gfx1250, wave32) ----
typedef __attribute__((ext_vector_type(16))) __bf16 v16bf;
typedef __attribute__((ext_vector_type(8)))  __bf16 v8bf;
typedef __attribute__((ext_vector_type(8)))  float  v8f;

// ---- problem constants ----
#define C_IN   256
#define C_OUT  256
#define H_DIM  56
#define W_DIM  56
#define HW     (H_DIM * W_DIM)          // 3136
#define KTOT   (C_IN * 9)               // 2304
#define NTOT   (32 * HW)                // 100352
#define NSTEPS 72                       // KTOT / BK

// ---- tile config: 8 waves (256 thr) per WG; 128co x 64n output block ----
#define BN 64
#define BK 32
#define BM 128
#define BK_PAD (BK + 16)                // 48 bf16 = 96B row stride (32B-aligned frags)

// LDS-only workgroup barrier: waits own DS ops, signals, waits. Unlike
// __syncthreads() it does NOT drain loadcnt, so prefetched global loads
// stay in flight across the barrier.  (CDNA5 ISA)
#define WG_BARRIER_LDS() asm volatile(            \
    "s_wait_dscnt 0x0\n\t"                        \
    "s_barrier_signal -1\n\t"                     \
    "s_barrier_wait -1" ::: "memory")

// truncation split: x = hi(bf16) + lo(bf16) + O(2^-16 * x)
__device__ __forceinline__ void trunc_split(float v, unsigned& hb, unsigned& lb)
{
    const unsigned xb = __float_as_uint(v);
    hb = xb >> 16;
    const float hif = __uint_as_float(xb & 0xFFFF0000u);
    lb = __float_as_uint(v - hif) >> 16;
}

// =====================================================================
// Pre-pass: split fp32 weights into hi/lo bf16 AND transpose the K axis
// from (ci*9 + r) to (r*256 + ci).  dst flat = co*2304 + r*256 + ci
// =====================================================================
__global__ __launch_bounds__(256)
void pack_weights(const float* __restrict__ w,
                  __bf16* __restrict__ whi,
                  __bf16* __restrict__ wlo)
{
    const int idx = blockIdx.x * 256 + threadIdx.x;   // over 256*2304
    const int co  = idx / KTOT;
    const int rem = idx - co * KTOT;
    const int r   = rem >> 8;          // 0..8
    const int ci  = rem & 255;
    const float v = w[(size_t)co * KTOT + ci * 9 + r];
    unsigned hb, lb;
    trunc_split(v, hb, lb);
    ((unsigned short*)whi)[idx] = (unsigned short)hb;
    ((unsigned short*)wlo)[idx] = (unsigned short)lb;
}

// =====================================================================
// Main kernel: implicit-GEMM conv via bf16x3 WMMA.
//   M = co (256), N = b*h*w (100352), K = s*32 (2304)
// Double-buffered LDS B tile, LDS-only barrier, pipelined im2col gather.
// =====================================================================
__global__ __launch_bounds__(256)
void conv3x3_wmma_bf16x3(const float* __restrict__ x,
                         const __bf16* __restrict__ whi,
                         const __bf16* __restrict__ wlo,
                         float* __restrict__ out)
{
    __shared__ __bf16 sBhi[2][BN][BK_PAD];
    __shared__ __bf16 sBlo[2][BN][BK_PAD];

    const int tid  = threadIdx.x;
    const int lane = tid & 31;
    const int wave = tid >> 5;
    const int wm   = wave & 3;            // co quadrant (32 co each)
    const int wn   = wave >> 2;           // n half (32 n each)

    const int n_block  = blockIdx.x * BN;
    const int co_block = blockIdx.y * BM;

    // ---- staging thread mapping: 64 n-columns x 4 k-groups of 8 ----
    const int s_n  = tid & 63;
    const int s_k0 = (tid >> 6) * 8;

    const int sn_g = n_block + s_n;       // 64-blocks never straddle a batch image
    const int s_b  = sn_g / HW;
    const int s_hw = sn_g - s_b * HW;
    const int s_ho = s_hw / W_DIM;
    const int s_wo = s_hw - s_ho * W_DIM;
    const float* __restrict__ ximg = x + (size_t)s_b * C_IN * HW;

    // ---- per-lane fragment geometry (contiguous k = 16*(lane/16) + s for A and B) ----
    const int frag_n = wn * 32 + (lane & 15);        // first B column row
    const int kbase  = (lane < 16) ? 0 : 16;         // B k-half
    const int asel   = (lane < 16) ? 0 : 16;         // A k-half select
    const int co_r0  = co_block + wm * 32 + (lane & 15);   // first co tile row
    const __bf16* __restrict__ wh0 = whi + (size_t)co_r0 * KTOT;
    const __bf16* __restrict__ wl0 = wlo + (size_t)co_r0 * KTOT;
    const __bf16* __restrict__ wh1 = wh0 + (size_t)16 * KTOT;  // second co tile
    const __bf16* __restrict__ wl1 = wl0 + (size_t)16 * KTOT;

    v8f c00 = {}, c01 = {}, c10 = {}, c11 = {};

    // ---- filter-tap geometry, advanced incrementally (no divisions) ----
    int kh = 0, kw = -1;
    float vmask = 0.0f;
    const float* __restrict__ xbase = ximg;
    auto advance_tap = [&]() {
        if (++kw == 3) { kw = 0; ++kh; }
        const int ih = s_ho + kh - 1;
        const int iw = s_wo + kw - 1;
        const bool valid = ((unsigned)ih < (unsigned)H_DIM) &
                           ((unsigned)iw < (unsigned)W_DIM);
        vmask = valid ? 1.0f : 0.0f;
        const int ihc = min(max(ih, 0), H_DIM - 1);   // always-legal address
        const int iwc = min(max(iw, 0), W_DIM - 1);
        xbase = ximg + ihc * W_DIM + iwc;             // + ci*HW
    };

    // prime the pipeline: gather for step 0 (tap r=0, ci0=0)
    advance_tap();
    float v[8];
    #pragma unroll
    for (int j = 0; j < 8; ++j)
        v[j] = xbase[(size_t)(s_k0 + j) * HW] * vmask;

    for (int s = 0; s < NSTEPS; ++s) {
        const int p = s & 1;

        // ---- A fragments first: pure-global loads overlap the barrier ----
        const int koff = s << 5;
        #define LOAD_AFRAG(P) __builtin_shufflevector(                    \
            *(const v8bf*)((P) + koff + asel),                            \
            *(const v8bf*)((P) + koff + 8 + asel),                        \
            0,1,2,3,4,5,6,7,8,9,10,11,12,13,14,15)
        const v16bf ah0 = LOAD_AFRAG(wh0);
        const v16bf al0 = LOAD_AFRAG(wl0);
        const v16bf ah1 = LOAD_AFRAG(wh1);
        const v16bf al1 = LOAD_AFRAG(wl1);
        #undef LOAD_AFRAG

        // ---- split current gather + store to LDS buffer p ----
        unsigned hb[8], lb[8];
        #pragma unroll
        for (int j = 0; j < 8; ++j) trunc_split(v[j], hb[j], lb[j]);
        const uint4 hv = { hb[0] | (hb[1] << 16), hb[2] | (hb[3] << 16),
                           hb[4] | (hb[5] << 16), hb[6] | (hb[7] << 16) };
        const uint4 lv = { lb[0] | (lb[1] << 16), lb[2] | (lb[3] << 16),
                           lb[4] | (lb[5] << 16), lb[6] | (lb[7] << 16) };
        *(uint4*)&sBhi[p][s_n][s_k0] = hv;
        *(uint4*)&sBlo[p][s_n][s_k0] = lv;

        // ---- prefetch next step's gather (spans barrier + WMMAs) ----
        if (s + 1 < NSTEPS) {
            const int ci0n = ((s + 1) & 7) << 5;
            if (ci0n == 0) advance_tap();
            #pragma unroll
            for (int j = 0; j < 8; ++j)
                v[j] = xbase[(size_t)(ci0n + s_k0 + j) * HW] * vmask;
        }

        // LDS-only barrier: no loadcnt drain, prefetch stays in flight.
        // WAR on buf p is two barriers away; each wave's frag ds_loads are
        // covered by its own s_wait_dscnt 0 before signaling.
        WG_BARRIER_LDS();

        // ---- B fragments from LDS (contiguous 32B, 32B-aligned) ----
        const v16bf bh0 = *(const v16bf*)&sBhi[p][frag_n][kbase];
        const v16bf bl0 = *(const v16bf*)&sBlo[p][frag_n][kbase];
        const v16bf bh1 = *(const v16bf*)&sBhi[p][frag_n + 16][kbase];
        const v16bf bl1 = *(const v16bf*)&sBlo[p][frag_n + 16][kbase];

        // ---- bf16x3 on a 2x2 register tile (A and B reused 6x each) ----
        #define WMMA(C,A,B) C = __builtin_amdgcn_wmma_f32_16x16x32_bf16( \
            false, A, false, B, (short)0, C, false, false)
        WMMA(c00, ah0, bh0); WMMA(c00, ah0, bl0); WMMA(c00, al0, bh0);
        WMMA(c01, ah0, bh1); WMMA(c01, ah0, bl1); WMMA(c01, al0, bh1);
        WMMA(c10, ah1, bh0); WMMA(c10, ah1, bl0); WMMA(c10, al1, bh0);
        WMMA(c11, ah1, bh1); WMMA(c11, ah1, bl1); WMMA(c11, al1, bh1);
        #undef WMMA
    }

    // ---- store D tiles: VGPR q -> M = q + 8*(lane>=16), N = lane&15 ----
    const int n_out = n_block + wn * 32 + (lane & 15);
    const int ob    = n_out / HW;
    const int ohw   = n_out - ob * HW;
    const int co0   = co_block + wm * 32 + ((lane < 16) ? 0 : 8);
    float* __restrict__ ob0 = out + (size_t)ob * C_OUT * HW + ohw;        // n tile 0
    float* __restrict__ ob1 = ob0 + 16;                                   // n tile 1
    #pragma unroll
    for (int q = 0; q < 8; ++q) {
        ob0[(size_t)(co0 + q) * HW]      = c00[q];
        ob1[(size_t)(co0 + q) * HW]      = c01[q];
        ob0[(size_t)(co0 + 16 + q) * HW] = c10[q];
        ob1[(size_t)(co0 + 16 + q) * HW] = c11[q];
    }
}

extern "C" void kernel_launch(void* const* d_in, const int* in_sizes, int n_in,
                              void* d_out, int out_size, void* d_ws, size_t ws_size,
                              hipStream_t stream)
{
    const float* x = (const float*)d_in[0];
    const float* w = (const float*)d_in[1];
    float* out = (float*)d_out;

    // workspace: hi + lo bf16 weights, [co][r][ci] layout (2.36 MB total)
    __bf16* whi = (__bf16*)d_ws;
    __bf16* wlo = whi + (size_t)C_OUT * KTOT;

    pack_weights<<<(C_OUT * KTOT) / 256, 256, 0, stream>>>(w, whi, wlo);

    dim3 grid(NTOT / BN, C_OUT / BM);   // (1568, 2)
    conv3x3_wmma_bf16x3<<<grid, 256, 0, stream>>>(x, whi, wlo, out);
}
